// OrthoLinear_90125593740101
// MI455X (gfx1250) — matrix operand order repaired and checked by reading, not compiled, not run on hardware
//
#include <hip/hip_runtime.h>

typedef _Float16 half_t;
typedef _Float16 v8h  __attribute__((ext_vector_type(8)));
typedef _Float16 v16h __attribute__((ext_vector_type(16)));
typedef float    v8f  __attribute__((ext_vector_type(8)));
typedef int      v4i  __attribute__((ext_vector_type(4)));

#define IN_F  4096
#define OUT_F 4096

#define BM 128
#define BN 128
#define BK 64
#define LDSK 72   // padded LDS row stride in halves (64 data + 8 pad)

// --- CDNA5 async global->LDS copy (guarded; falls back to reg staging) ----
#if defined(__AMDGCN__) && __has_builtin(__builtin_amdgcn_global_load_async_to_lds_b128)
#define USE_ASYNC_LDS 1
typedef __attribute__((address_space(1))) v4i gv4i;   // global int4
typedef __attribute__((address_space(3))) v4i lv4i;   // LDS int4
static __device__ __forceinline__ void async_cp16(const half_t* g, half_t* l) {
    // build exact-addrspace pointers via integer casts:
    //  - as1 pointer is the plain 64-bit VA
    //  - generic LDS pointer keeps the LDS offset in addr[31:0] (ISA aperture rule)
    gv4i* gp = (gv4i*)(unsigned long long)(uintptr_t)g;
    lv4i* lp = (lv4i*)(unsigned int)(uintptr_t)l;
    __builtin_amdgcn_global_load_async_to_lds_b128(gp, lp, 0, 0);
}
static __device__ __forceinline__ void wait_async0() {
#if __has_builtin(__builtin_amdgcn_s_wait_asynccnt)
    __builtin_amdgcn_s_wait_asynccnt(0);
#else
    asm volatile("s_wait_asynccnt 0x0" ::: "memory");
#endif
}
#else
#define USE_ASYNC_LDS 0
#endif

// ---------------------------------------------------------------------------
// Kernel 1: dequantize one W row (int4 packed -> f32 in LDS), scatter CSR
// outliers with ds_add_f32 atomics, then emit the row as f16.
// ---------------------------------------------------------------------------
__global__ __launch_bounds__(256) void dequant_w_kernel(
    const int*    __restrict__ packed,   // [OUT_F][IN_F/2], one byte per int32
    const float*  __restrict__ scales,   // [OUT_F]
    const half_t* __restrict__ vals,     // [nnz] fp16
    const int*    __restrict__ idx,      // [nnz]
    const int*    __restrict__ ptr,      // [OUT_F+1]
    half_t*       __restrict__ W16)      // [OUT_F][IN_F]
{
    __shared__ float row[IN_F];          // 16 KB
    const int r   = blockIdx.x;
    const int tid = threadIdx.x;
    const float s = scales[r];
    const int* prow = packed + (size_t)r * (IN_F / 2);

    #pragma unroll
    for (int j = 0; j < 8; ++j) {
        const int i = tid * 8 + j;
        const int p = prow[i];
        row[2 * i]     = (float)((p & 0xF) - 8) * s;          // low nibble -> even
        row[2 * i + 1] = (float)(((p >> 4) & 0xF) - 8) * s;   // high nibble -> odd
    }
    __syncthreads();

    const int beg = ptr[r], end = ptr[r + 1];
    for (int t = beg + tid; t < end; t += 256) {
        atomicAdd(&row[idx[t]], (float)vals[t]);
    }
    __syncthreads();

    half_t* wrow = W16 + (size_t)r * IN_F;
    #pragma unroll
    for (int j = 0; j < 2; ++j) {
        const int c0 = tid * 16 + j * 8;
        v8h h;
        #pragma unroll
        for (int e = 0; e < 8; ++e) h[e] = (half_t)row[c0 + e];
        *(v8h*)(wrow + c0) = h;
    }
}

// ---------------------------------------------------------------------------
// Kernel 2: x fp32 -> f16 (8 elements / thread)
// ---------------------------------------------------------------------------
__global__ __launch_bounds__(256) void convert_x_kernel(
    const float* __restrict__ x, half_t* __restrict__ X16, long n)
{
    const long i = ((long)blockIdx.x * 256 + threadIdx.x) * 8;
    if (i + 8 <= n) {
        float4 a = *(const float4*)(x + i);
        float4 b = *(const float4*)(x + i + 4);
        v8h h;
        h[0] = (half_t)a.x; h[1] = (half_t)a.y; h[2] = (half_t)a.z; h[3] = (half_t)a.w;
        h[4] = (half_t)b.x; h[5] = (half_t)b.y; h[6] = (half_t)b.z; h[7] = (half_t)b.w;
        *(v8h*)(X16 + i) = h;
    }
}

// ---------------------------------------------------------------------------
// Kernel 3: C[M,N] = X16[M,K] * W16[N,K]^T  via v_wmma_f32_16x16x32_f16.
// Block tile 128x128, BK=64, double-buffered LDS (one barrier per k-step).
// 8 wave32 waves in a 2(M) x 4(N) grid; wave tile 64x32 = 4x2 accumulators;
// 16 WMMAs per k-step per wave.
// ---------------------------------------------------------------------------
__global__ __launch_bounds__(256) void gemm_kernel(
    const half_t* __restrict__ X,   // [M][K] row-major
    const half_t* __restrict__ W,   // [N][K] row-major
    float*        __restrict__ C)   // [M][N] row-major
{
    __shared__ half_t Xs[2][BM][LDSK];
    __shared__ half_t Ws[2][BN][LDSK];

    const int tid  = threadIdx.x;
    const int lane = tid & 31;
    const int wave = tid >> 5;       // 0..7
    const int wn   = wave & 3;       // N wave group (32 cols each)
    const int wm   = wave >> 2;      // M wave group (64 rows each)

    const int m0 = blockIdx.y * BM;
    const int n0 = blockIdx.x * BN;

    // cooperative tile load: 2 threads per row, 32 halves (64B = 4x b128) each
    const int ldRow = tid >> 1;          // 0..127
    const int ldCol = (tid & 1) * 32;    // 0 or 32
    const half_t* xg = X + (size_t)(m0 + ldRow) * IN_F + ldCol;
    const half_t* wg = W + (size_t)(n0 + ldRow) * IN_F + ldCol;
    half_t* xsd = &Xs[0][ldRow][ldCol];
    half_t* wsd = &Ws[0][ldRow][ldCol];
    const int bufStride = BM * LDSK;     // halves per buffer

    // A fragment addressing (16-bit A 16x32 layout):
    //   lanes 0-15 : M=lane,    K in {0..7}  U {16..23}
    //   lanes 16-31: M=lane-16, K in {8..15} U {24..31}
    const int amRow = lane & 15;
    const int akOff = (lane >> 4) * 8;
    // B fragment addressing (16-bit B 32x16 layout):
    //   lanes 0-15 : N=lane, K in {0..7} U {16..23} ; lanes 16-31: N=lane-16, K in {8..15} U {24..31}
    const int bnRow = lane & 15;
    const int bk0   = (lane >> 4) * 8;

    v8f acc[4][2] = {};
#if !USE_ASYNC_LDS
    uint4 xr[4], wr[4];
#endif

    // ---- preload stage 0 into buffer 0 ----
#if USE_ASYNC_LDS
    #pragma unroll
    for (int c = 0; c < 4; ++c) {
        async_cp16(xg + c * 8, xsd + c * 8);
        async_cp16(wg + c * 8, wsd + c * 8);
    }
    wait_async0();
#else
    #pragma unroll
    for (int c = 0; c < 4; ++c) {
        xr[c] = *(const uint4*)(xg + c * 8);
        wr[c] = *(const uint4*)(wg + c * 8);
    }
    #pragma unroll
    for (int c = 0; c < 4; ++c) {
        *(uint4*)(xsd + c * 8) = xr[c];
        *(uint4*)(wsd + c * 8) = wr[c];
    }
#endif
    __syncthreads();

    const int NSTEPS = IN_F / BK;   // 64
    for (int s = 0; s < NSTEPS; ++s) {
        const int buf = s & 1;
        const int nxt = buf ^ 1;

        // ---- issue loads for stage s+1 (overlap with compute) ----
        if (s + 1 < NSTEPS) {
            const int kn = (s + 1) * BK;
#if USE_ASYNC_LDS
            #pragma unroll
            for (int c = 0; c < 4; ++c) {
                async_cp16(xg + kn + c * 8, xsd + nxt * bufStride + c * 8);
                async_cp16(wg + kn + c * 8, wsd + nxt * bufStride + c * 8);
            }
#else
            #pragma unroll
            for (int c = 0; c < 4; ++c) {
                xr[c] = *(const uint4*)(xg + kn + c * 8);
                wr[c] = *(const uint4*)(wg + kn + c * 8);
            }
#endif
        }

        // ---- compute stage s from buffer buf (two 32-wide k-slices) ----
        #pragma unroll
        for (int ks = 0; ks < 2; ++ks) {
            v16h a[4], b[2];
            #pragma unroll
            for (int i = 0; i < 4; ++i) {
                const half_t* p = &Xs[buf][wm * 64 + i * 16 + amRow][ks * 32 + akOff];
                union { v16h v; v8h h[2]; } u;
                u.h[0] = *(const v8h*)(p);        // K 0..7  / 8..15
                u.h[1] = *(const v8h*)(p + 16);   // K 16..23 / 24..31
                a[i] = u.v;
            }
            #pragma unroll
            for (int j = 0; j < 2; ++j) {
                const half_t* p = &Ws[buf][wn * 32 + j * 16 + bnRow][ks * 32 + bk0];
                union { v16h v; v8h h[2]; } u;
                u.h[0] = *(const v8h*)(p);        // K 0..7   / 8..15
                u.h[1] = *(const v8h*)(p + 16);   // K 16..23 / 24..31
                b[j] = u.v;
            }
            #pragma unroll
            for (int i = 0; i < 4; ++i)
                #pragma unroll
                for (int j = 0; j < 2; ++j)
                    acc[i][j] = __builtin_amdgcn_wmma_f32_16x16x32_f16(
                        false, a[i], false, b[j], (short)0, acc[i][j], false, false);
        }

        // ---- commit stage s+1 into the other buffer ----
        if (s + 1 < NSTEPS) {
#if USE_ASYNC_LDS
            wait_async0();
#else
            #pragma unroll
            for (int c = 0; c < 4; ++c) {
                *(uint4*)(xsd + nxt * bufStride + c * 8) = xr[c];
                *(uint4*)(wsd + nxt * bufStride + c * 8) = wr[c];
            }
#endif
            __syncthreads();
        }
    }

    // Epilogue: C/D layout -> lanes 0-15: N=lane, M=r ; lanes 16-31: N=lane-16, M=r+8
    const int cn   = lane & 15;
    const int cmHi = (lane >> 4) * 8;
    #pragma unroll
    for (int i = 0; i < 4; ++i) {
        #pragma unroll
        for (int j = 0; j < 2; ++j) {
            const int n = n0 + wn * 32 + j * 16 + cn;
            #pragma unroll
            for (int r = 0; r < 8; ++r) {
                const int m = m0 + wm * 64 + i * 16 + r + cmHi;
                C[(size_t)m * OUT_F + n] = acc[i][j][r];
            }
        }
    }
}

// ---------------------------------------------------------------------------
extern "C" void kernel_launch(void* const* d_in, const int* in_sizes, int n_in,
                              void* d_out, int out_size, void* d_ws, size_t ws_size,
                              hipStream_t stream) {
    const float*  x      = (const float*)d_in[0];
    const int*    packed = (const int*)d_in[1];
    const float*  scales = (const float*)d_in[2];
    const half_t* vals   = (const half_t*)d_in[3];
    const int*    idx    = (const int*)d_in[4];
    const int*    ptr    = (const int*)d_in[5];
    float* out = (float*)d_out;

    // workspace: W16 (32 MB) then X16 (tokens*K*2 bytes)
    half_t* W16 = (half_t*)d_ws;
    half_t* X16 = W16 + (size_t)OUT_F * IN_F;

    const int tokens = in_sizes[0] / IN_F;   // 4096

    dequant_w_kernel<<<OUT_F, 256, 0, stream>>>(packed, scales, vals, idx, ptr, W16);

    const long nx = (long)tokens * IN_F;
    convert_x_kernel<<<(int)((nx / 8 + 255) / 256), 256, 0, stream>>>(x, X16, nx);

    dim3 grid(OUT_F / BN, tokens / BM);
    gemm_kernel<<<grid, 256, 0, stream>>>(X16, W16, out);
}
